// PartAwareSlotAttention_51797305590103
// MI455X (gfx1250) — matrix operand order repaired and checked by reading, not compiled, not run on hardware
//
#include <hip/hip_runtime.h>

#define BB   64
#define CC   2048
#define NN   196
#define DD   256
#define SS   8
#define HIDN 512
#define NT   13          // ceil(196/16)
#define SCL  0.0625f     // 256^-0.5
#define KEPS 1e-8f
#define VKS  7           // ceil(224/32) k-steps for the updates GEMM

typedef __bf16 bf16;
typedef bf16  v16bf __attribute__((ext_vector_type(16)));
typedef bf16  v8bf  __attribute__((ext_vector_type(8)));
typedef float v8f   __attribute__((ext_vector_type(8)));

static __device__ inline v8f wmma_bf16(v16bf a, v16bf b, v8f c) {
  return __builtin_amdgcn_wmma_f32_16x16x32_bf16(false, a, false, b, (short)0, c,
                                                 false, false);
}

// ---- CDNA5 async global->LDS copies (ASYNCcnt-tracked) ----
static __device__ inline unsigned lds_off(const void* p) {
  return (unsigned)(size_t)p;  // LDS aperture: low 32 bits = LDS offset
}
static __device__ inline void async_b32(unsigned lds, const void* g) {
  asm volatile("global_load_async_to_lds_b32 %0, %1, off" :: "v"(lds), "v"(g) : "memory");
}
static __device__ inline void async_b128(unsigned lds, const void* g) {
  asm volatile("global_load_async_to_lds_b128 %0, %1, off" :: "v"(lds), "v"(g) : "memory");
}
static __device__ inline void wait_async0() {
  asm volatile("s_wait_asynccnt 0x0" ::: "memory");
}

// Build an A fragment (16x32, bf16) from an LDS f32 matrix with <=8 valid rows.
// lane L: m=L&15, half=L>>4; elem e -> K=16*half+e
static __device__ inline v16bf afrag8(const float* A, int lda, int k0, int lane) {
  const int hx = lane >> 4, m = lane & 15, mr = m & 7;
  const float sel = (m < 8) ? 1.f : 0.f;
  const float* p0 = A + mr * lda + k0 + 16 * hx;
  const float* p1 = p0 + 8;
  v16bf a;
#pragma unroll
  for (int e = 0; e < 8; ++e) {
    a[e]     = (bf16)(sel * p0[e]);
    a[e + 8] = (bf16)(sel * p1[e]);
  }
  return a;
}

// Swizzled-weight fragment: per (kstep, coltile) block of 512 bf16, lane L's 16
// values live contiguously at blk*512 + L*16  -> one 32B load.
static __device__ inline v16bf bfrag_sw(const bf16* __restrict__ Wsw, int ks,
                                        int ntiles, int tile, int lane) {
  return *(const v16bf*)(Wsw + (((size_t)ks * ntiles + tile) << 9) + lane * 16);
}

// Generic small GEMM: Out[8 x NC] (LDS f32) = A[8 x K] (LDS f32) @ Wsw (swizzled bf16) + bias
// mode 0: store; mode 1: relu+store; mode 2: Out += val
static __device__ inline void gemm_smallW(const float* A, int K,
                                          const bf16* __restrict__ Wsw,
                                          const float* __restrict__ bias, int NC,
                                          float* Out, int mode, int wid, int lane) {
  const int hx = lane >> 4, col = lane & 15;
  const int ntiles = NC >> 4;
  for (int tile = wid; tile < ntiles; tile += 8) {
    const int col0 = tile * 16;
    v8f acc = {};
    for (int k0 = 0; k0 < K; k0 += 32) {
      v16bf a = afrag8(A, K, k0, lane);
      v16bf bfr = bfrag_sw(Wsw, k0 >> 5, ntiles, tile, lane);
      acc = wmma_bf16(a, bfr, acc);
    }
    if (hx == 0) {  // rows 0..7 live in half 0
#pragma unroll
      for (int r = 0; r < 8; ++r) {
        float v = acc[r] + bias[col0 + col];
        if (mode == 1) v = fmaxf(v, 0.f);
        float* o = &Out[r * NC + col0 + col];
        if (mode == 2) *o += v; else *o = v;
      }
    }
  }
}

// LayerNorm over 8 rows x 256: wave w handles row w.
static __device__ inline void ln_rows8(const float (*X)[DD], float (*Y)[DD],
                                       const float* __restrict__ g,
                                       const float* __restrict__ bb,
                                       int wid, int lane) {
  const int row = wid;
  float s = 0.f, ss = 0.f;
#pragma unroll
  for (int i = 0; i < 8; ++i) {
    float v = X[row][lane + 32 * i];
    s += v; ss += v * v;
  }
#pragma unroll
  for (int o = 16; o > 0; o >>= 1) { s += __shfl_xor(s, o); ss += __shfl_xor(ss, o); }
  float mean = s * (1.f / 256.f);
  float var  = ss * (1.f / 256.f) - mean * mean;
  float rstd = rsqrtf(var + 1e-5f);
#pragma unroll
  for (int i = 0; i < 8; ++i) {
    int d = lane + 32 * i;
    Y[row][d] = (X[row][d] - mean) * rstd * g[d] + bb[d];
  }
}

// ---------------- weight pre-swizzle f32 -> fragment-ready bf16 ----------------
__global__ __launch_bounds__(256)
void k_swz(const float* __restrict__ src, bf16* __restrict__ dst, int K, int NC) {
  int o = blockIdx.x * 256 + threadIdx.x;
  if (o >= K * NC) return;
  int ntiles = NC >> 4;
  int blk = o >> 9, r = o & 511;
  int lane = r >> 4, e = r & 15;
  int ks = blk / ntiles, tile = blk - ks * ntiles;
  int hx = lane >> 4, col = lane & 15;
  int Krow = ks * 32 + 16 * hx + e;
  int d = tile * 16 + col;
  dst[o] = (bf16)src[(size_t)Krow * NC + d];
}

__global__ __launch_bounds__(256)
void k_zero(bf16* __restrict__ d, int n) {
  int i = blockIdx.x * 256 + threadIdx.x;
  if (i < n) d[i] = (bf16)0.f;
}

// ---------------- Kernel 1: xt = x^T @ conv_w + conv_b + pos ; inp = LN(xt) (bf16) ----------------
// Async double-buffered A pipeline; B fragments are direct 32B loads from swizzled conv_w.
__global__ __launch_bounds__(256)
void k_proj(const float* __restrict__ x, const bf16* __restrict__ cwsw,
            const float* __restrict__ convb, const float* __restrict__ pos,
            const float* __restrict__ g, const float* __restrict__ bta,
            bf16* __restrict__ inp) {
  __shared__ float Ash[2][2][16][36];  // [buf][batch][m][kk], f32 (async can't convert)
  __shared__ float Xt[16][260];

  const int t = threadIdx.x;
  const int wid = t >> 5, lane = t & 31;
  const int hx = lane >> 4, mrow = lane & 15, col = lane & 15;
  const int n0 = blockIdx.x * 16, b0 = blockIdx.y * 2;
  const int d0w = wid * 32;

  // zero A buffers once: rows with n>=NN must read as 0 for every slab
  for (int idx = t; idx < 2 * 2 * 16 * 36; idx += 256) (&Ash[0][0][0][0])[idx] = 0.f;

  v8f acc[2][2] = {};  // [batch][ct]

  auto issue = [&](int s, int buf) {
    const int c0 = s * 32;
#pragma unroll
    for (int bb = 0; bb < 2; ++bb) {
#pragma unroll
      for (int i = 0; i < 2; ++i) {
        int idx = t + 256 * i;
        int m = idx & 15, kk = idx >> 4;
        int n = n0 + m;
        if (n < NN)
          async_b32(lds_off(&Ash[buf][bb][m][kk]),
                    x + (size_t)((b0 + bb) * CC + c0 + kk) * NN + n);
      }
    }
  };

  issue(0, 0);
  for (int s = 0; s < 64; ++s) {
    const int buf = s & 1;
    wait_async0();          // own wave's slab-s asyncs done
    __syncthreads();        // publish; everyone past compute(s-1)
    if (s + 1 < 64) issue(s + 1, buf ^ 1);

    v16bf bfr[2];
#pragma unroll
    for (int ct = 0; ct < 2; ++ct)
      bfr[ct] = bfrag_sw(cwsw, s, 16, wid * 2 + ct, lane);
#pragma unroll
    for (int bb = 0; bb < 2; ++bb) {
      v16bf a;
      const float* p0 = &Ash[buf][bb][mrow][16 * hx];
      const float* p1 = p0 + 8;
#pragma unroll
      for (int e = 0; e < 8; ++e) { a[e] = (bf16)p0[e]; a[e + 8] = (bf16)p1[e]; }
      acc[bb][0] = wmma_bf16(a, bfr[0], acc[bb][0]);
      acc[bb][1] = wmma_bf16(a, bfr[1], acc[bb][1]);
    }
  }

  for (int bb = 0; bb < 2; ++bb) {     // per-batch epilogue (Xt reused)
    __syncthreads();
#pragma unroll
    for (int ct = 0; ct < 2; ++ct) {
      int d = d0w + 16 * ct + col;
#pragma unroll
      for (int r = 0; r < 8; ++r) {
        int M = r + 8 * hx;
        int n = n0 + M;
        int nc = (n < NN) ? n : 0;
        Xt[M][d] = acc[bb][ct][r] + convb[d] + pos[(size_t)nc * DD + d];
      }
    }
    __syncthreads();
    for (int rr = 0; rr < 2; ++rr) {   // LN rows wid and wid+8
      int m = wid + 8 * rr;
      float s = 0.f, ss = 0.f;
#pragma unroll
      for (int i = 0; i < 8; ++i) { float v = Xt[m][lane + 32 * i]; s += v; ss += v * v; }
#pragma unroll
      for (int o = 16; o > 0; o >>= 1) { s += __shfl_xor(s, o); ss += __shfl_xor(ss, o); }
      float mean = s * (1.f / 256.f);
      float var  = ss * (1.f / 256.f) - mean * mean;
      float rstd = rsqrtf(var + 1e-5f);
      int n = n0 + m;
      if (n < NN) {
#pragma unroll
        for (int i = 0; i < 8; ++i) {
          int d = lane + 32 * i;
          float v = (Xt[m][d] - mean) * rstd * g[d] + bta[d];
          inp[(size_t)((b0 + bb) * NN + n) * DD + d] = (bf16)v;
        }
      }
    }
  }
}

// ---------------- Kernel 2: k = inp@kw+kb (row-major), v = inp@vw+vb (swizzled for updates) ----
__global__ __launch_bounds__(256)
void k_kv(const bf16* __restrict__ inp,
          const bf16* __restrict__ kwsw, const bf16* __restrict__ vwsw,
          const float* __restrict__ kbias, const float* __restrict__ vbias,
          bf16* __restrict__ kout, bf16* __restrict__ vsw) {
  __shared__ bf16 Ain[2][16][32];

  const int t = threadIdx.x;
  const int wid = t >> 5, lane = t & 31;
  const int hx = lane >> 4, mrow = lane & 15, col = lane & 15;
  const int n0 = blockIdx.x * 16, b = blockIdx.y;
  const int d0w = wid * 32;

  for (int idx = t; idx < 2 * 16 * 32; idx += 256) (&Ain[0][0][0])[idx] = (bf16)0.f;

  v8f ak0 = {}, ak1 = {}, av0 = {}, av1 = {};

  auto issue = [&](int s, int buf) {
    const int c0 = s * 32;
    if (t < 64) {                       // A: 16 rows x 64B, contiguous bf16
      int m = t >> 2, part = t & 3;
      int n = n0 + m;
      if (n < NN)
        async_b128(lds_off(&Ain[buf][m][part * 8]),
                   inp + (size_t)(b * NN + n) * DD + c0 + part * 8);
    }
  };

  issue(0, 0);
  for (int s = 0; s < 8; ++s) {
    const int buf = s & 1;
    wait_async0();
    __syncthreads();
    if (s + 1 < 8) issue(s + 1, buf ^ 1);

    v16bf a;
    {
      const v8bf lo = *(const v8bf*)&Ain[buf][mrow][16 * hx];
      const v8bf hi = *(const v8bf*)&Ain[buf][mrow][16 * hx + 8];
#pragma unroll
      for (int e = 0; e < 8; ++e) { a[e] = lo[e]; a[e + 8] = hi[e]; }
    }
#pragma unroll
    for (int ct = 0; ct < 2; ++ct) {
      v16bf bk = bfrag_sw(kwsw, s, 16, wid * 2 + ct, lane);
      v16bf bv = bfrag_sw(vwsw, s, 16, wid * 2 + ct, lane);
      if (ct == 0) { ak0 = wmma_bf16(a, bk, ak0); av0 = wmma_bf16(a, bv, av0); }
      else         { ak1 = wmma_bf16(a, bk, ak1); av1 = wmma_bf16(a, bv, av1); }
    }
  }

  bf16* vswb = vsw + (size_t)b * VKS * 16 * 512;
#pragma unroll
  for (int ct = 0; ct < 2; ++ct) {
    const v8f ak = ct ? ak1 : ak0;
    const v8f av = ct ? av1 : av0;
    int d = d0w + 16 * ct + col;
#pragma unroll
    for (int r = 0; r < 8; ++r) {
      int M = r + 8 * hx;
      int n = n0 + M;
      if (n < NN) {
        kout[(size_t)(b * NN + n) * DD + d] = (bf16)(ak[r] + kbias[d]);
        // v stored fragment-ready for the updates GEMM:
        // blk=(n>>5)*16 + (d>>4); lane2=((n>>4)&1)*16 + (d&15); e=n&15
        size_t vo = (((size_t)(n >> 5) * 16 + (d >> 4)) << 9) +
                    ((((n >> 4) & 1) * 16 + (d & 15)) << 4) + (n & 15);
        vswb[vo] = (bf16)(av[r] + vbias[d]);
      }
    }
  }
}

// ---------------- Kernel 3: slot-attention iterations, one workgroup per batch ----------------
__global__ __launch_bounds__(256)
void k_slots(const float* __restrict__ noise, const float* __restrict__ mu,
             const float* __restrict__ logsig,
             const bf16* __restrict__ qwsw, const float* __restrict__ qb,
             const bf16* __restrict__ kbuf, const bf16* __restrict__ vsw,
             const bf16* __restrict__ wihsw, const bf16* __restrict__ whhsw,
             const float* __restrict__ bih, const float* __restrict__ bhh,
             const bf16* __restrict__ w1sw, const float* __restrict__ b1,
             const bf16* __restrict__ w2sw, const float* __restrict__ b2,
             const float* __restrict__ lnsg, const float* __restrict__ lnsb,
             const float* __restrict__ lnfg, const float* __restrict__ lnfb,
             float* __restrict__ out_slots, float* __restrict__ out_attn) {
  __shared__ float sl[SS][DD];      // slots
  __shared__ float sprev[SS][DD];
  __shared__ float sN[SS][DD];      // normalized slots
  __shared__ float qsu[SS][DD];     // q, later reused as updates
  __shared__ float attn[SS][224];   // padded K for updates GEMM
  __shared__ float pool[2 * SS * 768];  // gi|gh, later reused as ffh
  float (*gi)[768]  = (float(*)[768])pool;
  float (*gh)[768]  = (float(*)[768])(pool + SS * 768);
  float (*ffh)[HIDN] = (float(*)[HIDN])pool;

  const int t = threadIdx.x, b = blockIdx.x;
  const int wid = t >> 5, lane = t & 31;
  const int hx = lane >> 4, col = lane & 15;

  for (int idx = t; idx < SS * DD; idx += 256) {
    int s = idx >> 8, d = idx & 255;
    sl[s][d] = mu[d] + expf(logsig[d]) * noise[(size_t)(b * SS + s) * DD + d];
  }
  for (int idx = t; idx < SS * 224; idx += 256) (&attn[0][0])[idx] = 0.f;
  __syncthreads();

  for (int it = 0; it < 3; ++it) {
    for (int idx = t; idx < SS * DD; idx += 256)
      (&sprev[0][0])[idx] = (&sl[0][0])[idx];
    __syncthreads();

    ln_rows8(sl, sN, lnsg, lnsb, wid, lane);
    __syncthreads();

    // q = LN(slots) @ qw + qb
    gemm_smallW(&sN[0][0], DD, qwsw, qb, DD, &qsu[0][0], 0, wid, lane);
    __syncthreads();

    // dots[s][n] = scale * sum_d q[s][d] * k[n][d]  (B operand = k^T, contiguous bf16 per lane)
    {
      const bf16* kB = kbuf + (size_t)b * NN * DD;
      for (int tile = wid; tile < NT; tile += 8) {
        int n = tile * 16 + col;
        int nc = (n < NN) ? n : NN - 1;
        const bf16* krow = kB + (size_t)nc * DD + 16 * hx;
        v8f acc = {};
        for (int k0 = 0; k0 < DD; k0 += 32) {
          v16bf a = afrag8(&qsu[0][0], DD, k0, lane);
          v16bf bfr;
          const v8bf b0 = *(const v8bf*)(krow + k0);
          const v8bf b1 = *(const v8bf*)(krow + k0 + 8);
#pragma unroll
          for (int e = 0; e < 8; ++e) { bfr[e] = b0[e]; bfr[e + 8] = b1[e]; }
          acc = wmma_bf16(a, bfr, acc);
        }
        if (hx == 0 && n < NN) {
#pragma unroll
          for (int r = 0; r < 8; ++r) attn[r][n] = acc[r] * SCL;
        }
      }
    }
    __syncthreads();

    // softmax over slots (per key column) + EPS
    if (t < NN) {
      float mx = -1e30f;
#pragma unroll
      for (int s = 0; s < SS; ++s) mx = fmaxf(mx, attn[s][t]);
      float ex[SS]; float sum = 0.f;
#pragma unroll
      for (int s = 0; s < SS; ++s) { ex[s] = expf(attn[s][t] - mx); sum += ex[s]; }
      float inv = 1.f / sum;
#pragma unroll
      for (int s = 0; s < SS; ++s) attn[s][t] = ex[s] * inv + KEPS;
    }
    __syncthreads();

    // renormalize over keys (per slot row); wave wid handles row wid
    {
      int row = wid;
      float s = 0.f;
      for (int n = lane; n < NN; n += 32) s += attn[row][n];
#pragma unroll
      for (int o = 16; o > 0; o >>= 1) s += __shfl_xor(s, o);
      float inv = 1.f / s;
      for (int n = lane; n < NN; n += 32) attn[row][n] *= inv;
    }
    __syncthreads();

    // updates[s][d] = sum_n attn[s][n] * v[n][d]  (v pre-swizzled; pads zero)
    {
      const bf16* vB = vsw + (size_t)b * VKS * 16 * 512;
      for (int tile = wid; tile < (DD >> 4); tile += 8) {
        int d = tile * 16 + col;
        v8f acc = {};
        for (int k0 = 0; k0 < 224; k0 += 32) {
          v16bf a = afrag8(&attn[0][0], 224, k0, lane);
          v16bf bfr = bfrag_sw(vB, k0 >> 5, 16, tile, lane);
          acc = wmma_bf16(a, bfr, acc);
        }
        if (hx == 0) {
#pragma unroll
          for (int r = 0; r < 8; ++r) qsu[r][d] = acc[r];
        }
      }
    }
    __syncthreads();

    // GRU gate pre-activations
    gemm_smallW(&qsu[0][0],   DD, wihsw, bih, 768, &gi[0][0], 0, wid, lane);
    gemm_smallW(&sprev[0][0], DD, whhsw, bhh, 768, &gh[0][0], 0, wid, lane);
    __syncthreads();

    // GRU elementwise
    for (int idx = t; idx < SS * DD; idx += 256) {
      int s = idx >> 8, d = idx & 255;
      float ir = gi[s][d],       hr = gh[s][d];
      float iz = gi[s][DD + d],  hz = gh[s][DD + d];
      float in_ = gi[s][2*DD + d], hn = gh[s][2*DD + d];
      float r = 1.f / (1.f + expf(-(ir + hr)));
      float z = 1.f / (1.f + expf(-(iz + hz)));
      float nn = tanhf(in_ + r * hn);
      sl[s][d] = (1.f - z) * nn + z * sprev[s][d];
    }
    __syncthreads();

    // MLP with residual
    ln_rows8(sl, sN, lnfg, lnfb, wid, lane);
    __syncthreads();
    gemm_smallW(&sN[0][0], DD, w1sw, b1, HIDN, &ffh[0][0], 1, wid, lane);   // relu
    __syncthreads();
    gemm_smallW(&ffh[0][0], HIDN, w2sw, b2, DD, &sl[0][0], 2, wid, lane);   // accumulate
    __syncthreads();
  }

  for (int idx = t; idx < SS * DD; idx += 256)
    out_slots[(size_t)b * SS * DD + idx] = (&sl[0][0])[idx];
  for (int idx = t; idx < SS * NN; idx += 256) {
    int s = idx / NN, n = idx - s * NN;
    out_attn[(size_t)b * SS * NN + idx] = attn[s][n];
  }
}

// ---------------- host launcher ----------------
extern "C" void kernel_launch(void* const* d_in, const int* in_sizes, int n_in,
                              void* d_out, int out_size, void* d_ws, size_t ws_size,
                              hipStream_t stream) {
  (void)in_sizes; (void)n_in; (void)out_size; (void)ws_size;
  const float* x      = (const float*)d_in[0];
  const float* noise  = (const float*)d_in[1];
  const float* convw  = (const float*)d_in[2];
  const float* convb  = (const float*)d_in[3];
  const float* pos    = (const float*)d_in[4];
  const float* mu     = (const float*)d_in[5];
  const float* logsig = (const float*)d_in[6];
  const float* qw  = (const float*)d_in[7];  const float* qb  = (const float*)d_in[8];
  const float* kw  = (const float*)d_in[9];  const float* kb  = (const float*)d_in[10];
  const float* vw  = (const float*)d_in[11]; const float* vb  = (const float*)d_in[12];
  const float* wih = (const float*)d_in[13]; const float* whh = (const float*)d_in[14];
  const float* bih = (const float*)d_in[15]; const float* bhh = (const float*)d_in[16];
  const float* w1  = (const float*)d_in[17]; const float* b1  = (const float*)d_in[18];
  const float* w2  = (const float*)d_in[19]; const float* b2  = (const float*)d_in[20];
  const float* lnig = (const float*)d_in[21]; const float* lnib = (const float*)d_in[22];
  const float* lnsg = (const float*)d_in[23]; const float* lnsb = (const float*)d_in[24];
  const float* lnfg = (const float*)d_in[25]; const float* lnfb = (const float*)d_in[26];

  // workspace layout (bf16 elements; every block is a multiple of 512 -> 32B aligned)
  bf16* inp    = (bf16*)d_ws;                       // B*N*D
  bf16* kbuf   = inp    + (size_t)BB * NN * DD;     // row-major k
  bf16* vsw    = kbuf   + (size_t)BB * NN * DD;     // swizzled v: B * 7*16*512
  bf16* cw_sw  = vsw    + (size_t)BB * VKS * 16 * 512;
  bf16* kw_sw  = cw_sw  + (size_t)CC * DD;
  bf16* vw_sw  = kw_sw  + (size_t)DD * DD;
  bf16* qw_sw  = vw_sw  + (size_t)DD * DD;
  bf16* wih_sw = qw_sw  + (size_t)DD * DD;
  bf16* whh_sw = wih_sw + (size_t)DD * 3 * DD;
  bf16* w1_sw  = whh_sw + (size_t)DD * 3 * DD;
  bf16* w2_sw  = w1_sw  + (size_t)DD * HIDN;

  float* out_slots = (float*)d_out;                          // (B,S,D)
  float* out_attn  = out_slots + (size_t)BB * SS * DD;       // (B,S,N)

  auto swz = [&](const float* s, bf16* d, int K, int NC) {
    int n = K * NC;
    k_swz<<<(n + 255) / 256, 256, 0, stream>>>(s, d, K, NC);
  };
  swz(convw, cw_sw, CC, DD);
  swz(kw, kw_sw, DD, DD);
  swz(vw, vw_sw, DD, DD);
  swz(qw, qw_sw, DD, DD);
  swz(wih, wih_sw, DD, 3 * DD);
  swz(whh, whh_sw, DD, 3 * DD);
  swz(w1, w1_sw, DD, HIDN);
  swz(w2, w2_sw, HIDN, DD);
  {
    int n = BB * VKS * 16 * 512;
    k_zero<<<(n + 255) / 256, 256, 0, stream>>>(vsw, n);
  }

  k_proj<<<dim3(NT, BB / 2), 256, 0, stream>>>(x, cw_sw, convb, pos, lnig, lnib, inp);
  k_kv<<<dim3(NT, BB), 256, 0, stream>>>(inp, kw_sw, vw_sw, kb, vb, kbuf, vsw);
  k_slots<<<dim3(BB), 256, 0, stream>>>(noise, mu, logsig, qw_sw, qb, kbuf, vsw,
                                        wih_sw, whh_sw, bih, bhh, w1_sw, b1, w2_sw, b2,
                                        lnsg, lnsb, lnfg, lnfb, out_slots, out_attn);
}
